// Attention_Database_3246995275823
// MI455X (gfx1250) — matrix operand order repaired and checked by reading, not compiled, not run on hardware
//
#include <hip/hip_runtime.h>
#include <math.h>

// ---------------- problem constants (match reference) ----------------
#define BAT     2048
#define SEQ     8
#define DMODEL  512
#define NHEAD   4
#define DK      128
#define NLAYER  4
#define DFF     2048
#define OUTLEN  30000
#define TOK     (BAT * SEQ)          // 16384 token rows

typedef __attribute__((ext_vector_type(16))) __bf16 v16bf;
typedef __attribute__((ext_vector_type(8)))  float  v8f;

__device__ __forceinline__ unsigned short f2bf(float f) {
    unsigned int u = __float_as_uint(f);
    u += 0x7fffu + ((u >> 16) & 1u);       // round-to-nearest-even
    return (unsigned short)(u >> 16);
}

// ---------------- tiny elementwise kernels ----------------

__global__ void k_fill_neginf(float* __restrict__ out, long long n) {
    long long i = (long long)blockIdx.x * blockDim.x + threadIdx.x;
    if (i < n) out[i] = __int_as_float(0xff800000u);   // -inf
}

// transpose-convert: W[K][N] f32 -> out[N][K] bf16 (N-major for GEMM B tiles)
__global__ void k_cvtT(const float* __restrict__ W, unsigned short* __restrict__ out,
                       int K, int N) {
    int idx = blockIdx.x * blockDim.x + threadIdx.x;
    if (idx >= K * N) return;
    int k = idx / N, n = idx - k * N;
    out[(size_t)n * K + k] = f2bf(W[idx]);
}

// pack bq|bk|bv -> bqkv[L][1536]
__global__ void k_bias_pack(const float* __restrict__ bq, const float* __restrict__ bk,
                            const float* __restrict__ bv, float* __restrict__ bqkv) {
    int idx = blockIdx.x * blockDim.x + threadIdx.x;
    if (idx >= NLAYER * 3 * DMODEL) return;
    int L = idx / (3 * DMODEL), j = idx % (3 * DMODEL);
    float v;
    if (j < DMODEL)           v = bq[L * DMODEL + j];
    else if (j < 2 * DMODEL)  v = bk[L * DMODEL + j - DMODEL];
    else                      v = bv[L * DMODEL + j - 2 * DMODEL];
    bqkv[idx] = v;
}

// x[b,s,d] = emb[s][sentence[b,s]][d] + PE[s,d]
__global__ void k_embed(const int* __restrict__ sentence, const float* __restrict__ emb,
                        float* __restrict__ x) {
    long long idx = (long long)blockIdx.x * blockDim.x + threadIdx.x;
    if (idx >= (long long)TOK * DMODEL) return;
    int t = (int)(idx >> 9);           // token row (b*S+s)
    int d = (int)(idx & 511);
    int s = t & (SEQ - 1);
    int tok = sentence[t];
    int de = d & ~1;
    float div = __expf((float)de * (-9.210340371976184f / (float)DMODEL)); // -ln(1e4)/D
    float ang = (float)s * div;
    float pe = (d & 1) ? __cosf(ang) : __sinf(ang);
    x[idx] = emb[((size_t)s * OUTLEN + tok) * DMODEL + d] + pe;
}

// LayerNorm (unbiased std, torch-style): out_bf16 = a*(x-mu)/(std+eps)+b
// one wave per row, 8 rows per 256-thread block
__global__ __launch_bounds__(256) void k_ln(const float* __restrict__ x,
                                            const float* __restrict__ ga,
                                            const float* __restrict__ gb,
                                            unsigned short* __restrict__ out, int rows) {
    int wave = threadIdx.x >> 5, lane = threadIdx.x & 31;
    int row = blockIdx.x * 8 + wave;
    if (row >= rows) return;
    const float* xr = x + (size_t)row * DMODEL;
    float v[16];
#pragma unroll
    for (int i = 0; i < 16; i++) v[i] = xr[lane + i * 32];
    float s = 0.f;
#pragma unroll
    for (int i = 0; i < 16; i++) s += v[i];
#pragma unroll
    for (int m = 16; m >= 1; m >>= 1) s += __shfl_xor(s, m, 32);
    float mu = s * (1.f / (float)DMODEL);
    float q = 0.f;
#pragma unroll
    for (int i = 0; i < 16; i++) { float d = v[i] - mu; q += d * d; }
#pragma unroll
    for (int m = 16; m >= 1; m >>= 1) q += __shfl_xor(q, m, 32);
    float sd = sqrtf(q * (1.f / (float)(DMODEL - 1)));
    float inv = 1.f / (sd + 1e-6f);
    unsigned short* orow = out + (size_t)row * DMODEL;
#pragma unroll
    for (int i = 0; i < 16; i++) {
        int d = lane + i * 32;
        orow[d] = f2bf(ga[d] * (v[i] - mu) * inv + gb[d]);
    }
}

// final LN applied only to the selected (masked) token of each sample
__global__ __launch_bounds__(256) void k_ln_select(const float* __restrict__ x,
                                                   const float* __restrict__ ga,
                                                   const float* __restrict__ gb,
                                                   const int* __restrict__ midx,
                                                   unsigned short* __restrict__ out) {
    int wave = threadIdx.x >> 5, lane = threadIdx.x & 31;
    int b = blockIdx.x * 8 + wave;
    if (b >= BAT) return;
    int row = b * SEQ + midx[b];
    const float* xr = x + (size_t)row * DMODEL;
    float v[16];
#pragma unroll
    for (int i = 0; i < 16; i++) v[i] = xr[lane + i * 32];
    float s = 0.f;
#pragma unroll
    for (int i = 0; i < 16; i++) s += v[i];
#pragma unroll
    for (int m = 16; m >= 1; m >>= 1) s += __shfl_xor(s, m, 32);
    float mu = s * (1.f / (float)DMODEL);
    float q = 0.f;
#pragma unroll
    for (int i = 0; i < 16; i++) { float d = v[i] - mu; q += d * d; }
#pragma unroll
    for (int m = 16; m >= 1; m >>= 1) q += __shfl_xor(q, m, 32);
    float sd = sqrtf(q * (1.f / (float)(DMODEL - 1)));
    float inv = 1.f / (sd + 1e-6f);
    unsigned short* orow = out + (size_t)b * DMODEL;
#pragma unroll
    for (int i = 0; i < 16; i++) {
        int d = lane + i * 32;
        orow[d] = f2bf(ga[d] * (v[i] - mu) * inv + gb[d]);
    }
}

// tiny attention (S=8): one thread per (b, head, query). qkv packed [TOK][1536].
// reference quirk: mask broadcasts over the *query* index ([B,1,S,1] vs [B,h,q,k]).
__global__ void k_attn(const float* __restrict__ qkv, const float* __restrict__ mask,
                       unsigned short* __restrict__ attb) {
    int gid = blockIdx.x * blockDim.x + threadIdx.x;
    if (gid >= BAT * NHEAD * SEQ) return;
    int si = gid & 7, h = (gid >> 3) & 3, b = gid >> 5;
    const float scale = 0.08838834764831845f;   // 1/sqrt(128)
    bool mq = (mask[b * SEQ + si] == 0.f);
    const float* qrow = qkv + (size_t)(b * SEQ + si) * 1536 + h * DK;
    float sc[SEQ];
#pragma unroll
    for (int sj = 0; sj < SEQ; sj++) {
        const float* krow = qkv + (size_t)(b * SEQ + sj) * 1536 + DMODEL + h * DK;
        float s = 0.f;
        for (int d = 0; d < DK; d++) s += qrow[d] * krow[d];
        sc[sj] = mq ? -1e9f : s * scale;
    }
    float mx = sc[0];
#pragma unroll
    for (int j = 1; j < SEQ; j++) mx = fmaxf(mx, sc[j]);
    float p[SEQ], sum = 0.f;
#pragma unroll
    for (int j = 0; j < SEQ; j++) { p[j] = __expf(sc[j] - mx); sum += p[j]; }
    float inv = 1.f / sum;
#pragma unroll
    for (int j = 0; j < SEQ; j++) p[j] *= inv;
    unsigned short* orow = attb + (size_t)(b * SEQ + si) * DMODEL + h * DK;
    for (int d = 0; d < DK; d++) {
        float a = 0.f;
#pragma unroll
        for (int sj = 0; sj < SEQ; sj++)
            a += p[sj] * qkv[(size_t)(b * SEQ + sj) * 1536 + 2 * DMODEL + h * DK + d];
        orow[d] = f2bf(a);
    }
}

// ---------------- WMMA GEMM: C[M,N] = A[M,K](bf16) x B[N-major: [N][K]] + bias ----------------
// block = 256 threads = 8 waves, arranged 4 (M) x 2 (N).
// Block tile: M=128, N=128, K-step=32. Each wave owns a 32x64 tile:
//   2 A-fragments x 4 B-fragments -> 8 v_wmma_f32_16x16x32_bf16 per K-step,
// doubling WMMA work per ds_load_b128 and per barrier vs a 16x64 wave tile.

#define BM 128
#define BN 128
#define BK 32
#define LDSS 40   // LDS row stride in bf16 elems: 80 bytes -> 16B-aligned rows, low conflicts

template<bool BF32, bool RELU, bool RESID, bool OUTB16, bool PRED>
__global__ __launch_bounds__(256) void k_gemm(
    const unsigned short* __restrict__ A, const void* __restrict__ Bv,
    const float* __restrict__ bias, float* __restrict__ Cf,
    unsigned short* __restrict__ Cb, const float* __restrict__ resid,
    int M, int N, int K, int ldc,
    const int* __restrict__ midx, int kc) {

    __shared__ __align__(16) unsigned short As[BM][LDSS];
    __shared__ __align__(16) unsigned short Bs[BN][LDSS];

    const int tid    = threadIdx.x;
    const int m0     = blockIdx.y * BM;
    const int n0     = blockIdx.x * BN;
    const int wave   = tid >> 5;
    const int wave_m = wave & 3;          // 4 waves along M (32 rows each)
    const int wave_n = wave >> 2;         // 2 waves along N (64 cols each)
    const int lane   = tid & 31;
    const int lhalf  = lane >> 4;         // 0: lanes 0-15, 1: lanes 16-31
    const int l16    = lane & 15;

    const v8f vzero = {0.f, 0.f, 0.f, 0.f, 0.f, 0.f, 0.f, 0.f};
    v8f acc[8];                            // [mi][nsub] -> acc[mi*4+nsub]
#pragma unroll
    for (int i = 0; i < 8; i++) acc[i] = vzero;

    // cooperative-load roles: both tiles are 128 rows x 32 bf16 (16 elems/thread)
    const int ldrow  = tid >> 1;
    const int ldhalf = tid & 1;

    for (int k0 = 0; k0 < K; k0 += BK) {
        // ---- stage A tile (bf16, row-major [M][K]) ----
        {
            uint4 v0 = make_uint4(0, 0, 0, 0), v1 = make_uint4(0, 0, 0, 0);
            int gm = m0 + ldrow;
            if (gm < M) {
                const uint4* src = (const uint4*)(A + (size_t)gm * K + k0 + ldhalf * 16);
                v0 = src[0]; v1 = src[1];
            }
            uint4* dst = (uint4*)(&As[ldrow][ldhalf * 16]);
            dst[0] = v0; dst[1] = v1;
        }
        // ---- stage B tile (N-major [N][K]; f32 source converted in-flight) ----
        {
            int gn = n0 + ldrow;
            if (BF32) {
                float f[16];
#pragma unroll
                for (int j = 0; j < 16; j++) f[j] = 0.f;
                if (gn < N) {
                    const float4* src =
                        (const float4*)((const float*)Bv + (size_t)gn * K + k0 + ldhalf * 16);
#pragma unroll
                    for (int c = 0; c < 4; c++) {
                        float4 t = src[c];
                        f[c * 4 + 0] = t.x; f[c * 4 + 1] = t.y;
                        f[c * 4 + 2] = t.z; f[c * 4 + 3] = t.w;
                    }
                }
                union { uint4 u[2]; unsigned short s[16]; } pk;
#pragma unroll
                for (int j = 0; j < 16; j++) pk.s[j] = f2bf(f[j]);
                uint4* dst = (uint4*)(&Bs[ldrow][ldhalf * 16]);
                dst[0] = pk.u[0]; dst[1] = pk.u[1];
            } else {
                uint4 v0 = make_uint4(0, 0, 0, 0), v1 = make_uint4(0, 0, 0, 0);
                if (gn < N) {
                    const uint4* src =
                        (const uint4*)((const unsigned short*)Bv + (size_t)gn * K + k0 + ldhalf * 16);
                    v0 = src[0]; v1 = src[1];
                }
                uint4* dst = (uint4*)(&Bs[ldrow][ldhalf * 16]);
                dst[0] = v0; dst[1] = v1;
            }
        }
        __syncthreads();

        // ---- A fragments (16x32): lanes 0-15 -> K 0..7 | 16..23, lanes 16-31 -> K 8..15 | 24..31
        v16bf afr[2];
#pragma unroll
        for (int mi = 0; mi < 2; mi++) {
            uint4* ap = (uint4*)&afr[mi];
            const int arw = wave_m * 32 + mi * 16 + l16;
            const int ac  = lhalf * 8;
            ap[0] = *(const uint4*)(&As[arw][ac]);
            ap[1] = *(const uint4*)(&As[arw][ac + 16]);
        }
        // ---- B fragments: lane -> N col, lanes 0-15 -> K 0..7 | 16..23, lanes 16-31 -> K 8..15 | 24..31
        v16bf bfr[4];
#pragma unroll
        for (int nsub = 0; nsub < 4; nsub++) {
            uint4* bp = (uint4*)&bfr[nsub];
            const int brw = wave_n * 64 + nsub * 16 + l16;
            const int bc  = lhalf * 8;
            bp[0] = *(const uint4*)(&Bs[brw][bc]);
            bp[1] = *(const uint4*)(&Bs[brw][bc + 16]);
        }
        // ---- 8 WMMAs per K-step
#pragma unroll
        for (int mi = 0; mi < 2; mi++)
#pragma unroll
            for (int nsub = 0; nsub < 4; nsub++)
                acc[mi * 4 + nsub] = __builtin_amdgcn_wmma_f32_16x16x32_bf16(
                    false, afr[mi], false, bfr[nsub], (short)0, acc[mi * 4 + nsub],
                    false, false);
        __syncthreads();
    }

    // ---- epilogue: C VGPR r -> M = r (lanes 0-15) / r+8 (lanes 16-31), N = lane%16
#pragma unroll
    for (int mi = 0; mi < 2; mi++) {
#pragma unroll
        for (int nsub = 0; nsub < 4; nsub++) {
            int n = n0 + wave_n * 64 + nsub * 16 + l16;
            if (n >= N) continue;
            float bn = bias[n];
#pragma unroll
            for (int r = 0; r < 8; r++) {
                int m = m0 + wave_m * 32 + mi * 16 + r + lhalf * 8;
                if (m >= M) continue;
                float c = acc[mi * 4 + nsub][r] + bn;
                if (RELU)  c = fmaxf(c, 0.f);
                if (RESID) c += resid[(size_t)m * ldc + n];
                if (PRED) {
                    if (midx[m] == kc) Cf[(size_t)m * ldc + n] = c;
                } else if (OUTB16) {
                    Cb[(size_t)m * ldc + n] = f2bf(c);
                } else {
                    Cf[(size_t)m * ldc + n] = c;
                }
            }
        }
    }
}

// ---------------- host orchestration ----------------

extern "C" void kernel_launch(void* const* d_in, const int* in_sizes, int n_in,
                              void* d_out, int out_size, void* d_ws, size_t ws_size,
                              hipStream_t stream) {
    const int*   sentence = (const int*)  d_in[0];
    const float* mask     = (const float*)d_in[1];
    const int*   midx     = (const int*)  d_in[2];
    const float* emb      = (const float*)d_in[3];
    const float* dec_bias = (const float*)d_in[4];
    const float* Wq = (const float*)d_in[5],  *bq = (const float*)d_in[6];
    const float* Wk = (const float*)d_in[7],  *bk = (const float*)d_in[8];
    const float* Wv = (const float*)d_in[9],  *bv = (const float*)d_in[10];
    const float* Wo = (const float*)d_in[11], *bo = (const float*)d_in[12];
    const float* W1 = (const float*)d_in[13], *b1 = (const float*)d_in[14];
    const float* W2 = (const float*)d_in[15], *b2 = (const float*)d_in[16];
    const float* ln1a = (const float*)d_in[17], *ln1b = (const float*)d_in[18];
    const float* ln2a = (const float*)d_in[19], *ln2b = (const float*)d_in[20];
    const float* fna  = (const float*)d_in[21], *fnb  = (const float*)d_in[22];

    // workspace carve-up (256B aligned)
    size_t off = 0;
    auto carve = [&](size_t bytes) {
        size_t o = off;
        off = (off + bytes + 255) & ~(size_t)255;
        return (char*)d_ws + o;
    };
    float*          x    = (float*)         carve((size_t)TOK * DMODEL * 4);
    float*          qkv  = (float*)         carve((size_t)TOK * 3 * DMODEL * 4);
    unsigned short* hb   = (unsigned short*)carve((size_t)TOK * DMODEL * 2);
    unsigned short* attb = (unsigned short*)carve((size_t)TOK * DMODEL * 2);
    unsigned short* ffb  = (unsigned short*)carve((size_t)TOK * DFF * 2);
    unsigned short* hsel = (unsigned short*)carve((size_t)BAT * DMODEL * 2);
    unsigned short* qkvT = (unsigned short*)carve((size_t)NLAYER * 3 * DMODEL * DMODEL * 2);
    unsigned short* woT  = (unsigned short*)carve((size_t)NLAYER * DMODEL * DMODEL * 2);
    unsigned short* w1T  = (unsigned short*)carve((size_t)NLAYER * DMODEL * DFF * 2);
    unsigned short* w2T  = (unsigned short*)carve((size_t)NLAYER * DFF * DMODEL * 2);
    float*          bqkv = (float*)         carve((size_t)NLAYER * 3 * DMODEL * 4);
    (void)ws_size; (void)n_in; (void)in_sizes; (void)out_size;

    // 0) preds = -inf everywhere (pads + unwritten vocab tails)
    {
        long long n = (long long)BAT * OUTLEN;
        k_fill_neginf<<<dim3((unsigned)((n + 255) / 256)), 256, 0, stream>>>((float*)d_out, n);
    }

    // 1) weight convert+transpose to bf16 N-major; bias packing
    {
        int nDD = DMODEL * DMODEL, nDF = DMODEL * DFF;
        dim3 bDD((nDD + 255) / 256), bDF((nDF + 255) / 256);
        for (int L = 0; L < NLAYER; L++) {
            k_cvtT<<<bDD, 256, 0, stream>>>(Wq + (size_t)L * nDD,
                qkvT + (size_t)L * 3 * nDD + 0 * nDD, DMODEL, DMODEL);
            k_cvtT<<<bDD, 256, 0, stream>>>(Wk + (size_t)L * nDD,
                qkvT + (size_t)L * 3 * nDD + 1 * nDD, DMODEL, DMODEL);
            k_cvtT<<<bDD, 256, 0, stream>>>(Wv + (size_t)L * nDD,
                qkvT + (size_t)L * 3 * nDD + 2 * nDD, DMODEL, DMODEL);
            k_cvtT<<<bDD, 256, 0, stream>>>(Wo + (size_t)L * nDD,
                woT + (size_t)L * nDD, DMODEL, DMODEL);
            k_cvtT<<<bDF, 256, 0, stream>>>(W1 + (size_t)L * nDF,
                w1T + (size_t)L * nDF, DMODEL, DFF);       // [D][DFF] -> [DFF][D]
            k_cvtT<<<bDF, 256, 0, stream>>>(W2 + (size_t)L * nDF,
                w2T + (size_t)L * nDF, DFF, DMODEL);       // [DFF][D] -> [D][DFF]
        }
        int nb = NLAYER * 3 * DMODEL;
        k_bias_pack<<<(nb + 255) / 256, 256, 0, stream>>>(bq, bk, bv, bqkv);
    }

    // 2) embeddings + positional encoding
    {
        long long n = (long long)TOK * DMODEL;
        k_embed<<<dim3((unsigned)((n + 255) / 256)), 256, 0, stream>>>(sentence, emb, x);
    }

    // 3) encoder layers
    dim3 blk(256);
    for (int L = 0; L < NLAYER; L++) {
        // LN1 -> hb (bf16)
        k_ln<<<TOK / 8, blk, 0, stream>>>(x, ln1a + L * DMODEL, ln1b + L * DMODEL, hb, TOK);
        // fused QKV: [TOK,512] x [512,1536] -> qkv f32
        k_gemm<false, false, false, false, false>
            <<<dim3(3 * DMODEL / BN, TOK / BM), blk, 0, stream>>>(
                hb, qkvT + (size_t)L * 3 * DMODEL * DMODEL, bqkv + L * 3 * DMODEL,
                qkv, nullptr, nullptr, TOK, 3 * DMODEL, DMODEL, 3 * DMODEL, nullptr, 0);
        // attention -> attb (bf16)
        k_attn<<<(BAT * NHEAD * SEQ) / 256, blk, 0, stream>>>(qkv, mask, attb);
        // O-proj + residual: x += attb x Wo + bo
        k_gemm<false, false, true, false, false>
            <<<dim3(DMODEL / BN, TOK / BM), blk, 0, stream>>>(
                attb, woT + (size_t)L * DMODEL * DMODEL, bo + L * DMODEL,
                x, nullptr, x, TOK, DMODEL, DMODEL, DMODEL, nullptr, 0);
        // LN2 -> hb
        k_ln<<<TOK / 8, blk, 0, stream>>>(x, ln2a + L * DMODEL, ln2b + L * DMODEL, hb, TOK);
        // FFN1: relu(hb x W1 + b1) -> ffb (bf16)
        k_gemm<false, true, false, true, false>
            <<<dim3(DFF / BN, TOK / BM), blk, 0, stream>>>(
                hb, w1T + (size_t)L * DMODEL * DFF, b1 + L * DFF,
                nullptr, ffb, nullptr, TOK, DFF, DMODEL, DFF, nullptr, 0);
        // FFN2 + residual: x += ffb x W2 + b2   (K = 2048)
        k_gemm<false, false, true, false, false>
            <<<dim3(DMODEL / BN, TOK / BM), blk, 0, stream>>>(
                ffb, w2T + (size_t)L * DFF * DMODEL, b2 + L * DMODEL,
                x, nullptr, x, TOK, DMODEL, DFF, DMODEL, nullptr, 0);
    }

    // 4) final LN on the selected masked token -> hsel (bf16 [BAT,512])
    k_ln_select<<<BAT / 8, blk, 0, stream>>>(x, fna, fnb, midx, hsel);

    // 5) decode: per vocab column kc, logits = hsel x emb[kc,:vk]^T + dec_bias[kc],
    //    predicated store where maskedIndex[b] == kc. emb row-major == N-major B.
    static const int VOC[SEQ] = {30000, 20000, 10000, 8000, 5000, 30000, 20000, 10000};
    for (int kc = 0; kc < SEQ; kc++) {
        int vk = VOC[kc];
        k_gemm<true, false, false, false, true>
            <<<dim3((vk + BN - 1) / BN, BAT / BM), blk, 0, stream>>>(
                hsel, emb + (size_t)kc * OUTLEN * DMODEL, dec_bias + (size_t)kc * OUTLEN,
                (float*)d_out, nullptr, nullptr, BAT, vk, DMODEL, OUTLEN, midx, kc);
    }
}
